// RNN_37598143709474
// MI455X (gfx1250) — matrix operand order repaired and checked by reading, not compiled, not run on hardware
//
#include <hip/hip_runtime.h>

// ---------------- Types ----------------
typedef __attribute__((ext_vector_type(16))) __bf16       v16bf;
typedef __attribute__((ext_vector_type(8)))  float        v8f;
typedef __attribute__((ext_vector_type(4)))  unsigned int v4u;
typedef __attribute__((ext_vector_type(8)))  int          v8i;
typedef __attribute__((ext_vector_type(4)))  int          v4i;

union BF16Frag { v16bf v; v4u q[2]; unsigned u[8]; };

// ---------------- Problem constants ----------------
#define B_   64
#define T_   2048
#define I_   512
#define H_   1024
#define O_   512
#define KC   (H_ + I_)      // 1536 fused K  ([h | x_t])
#define KP   (KC + 8)       // 1544: padded row (LDS bank-conflict-free: 772 dw % 64 = 4)
#define NBLK 64             // blocks; each owns 16 h-columns
#define ROWS 48             // 16 r-rows, 16 z-rows, 16 n-rows
#define WSLAB (ROWS * KP)   // bf16 elements per per-block weight slab

// ---------------- Workspace layout (bytes) ----------------
#define OFF_BAR   0
#define OFF_HF32  256                                     // f32 h  [64][1024]
#define OFF_HBF   (OFF_HF32 + B_*H_*4)                    // bf16 h double buffer [2][64][1024]
#define OFF_WP    (OFF_HBF + 2*B_*H_*2)                   // packed weights [64][48][1544] bf16
#define OFF_WOUT  (OFF_WP + NBLK*WSLAB*2)                 // w_out bf16 [512][1024]
#define SMEM_RED  (4*4*32*8*4)                            // 16 KB reduction buffer
#define SMEM_BYTES (WSLAB*2 + SMEM_RED)                   // 148224 + 16384 = 164608

// ---------------- Helpers ----------------
__device__ inline unsigned bfround(unsigned u) {
  return u + 0x7FFFu + ((u >> 16) & 1u);                  // RNE for bf16 truncation
}
__device__ inline unsigned short f2bf(float f) {
  return (unsigned short)(bfround(__builtin_bit_cast(unsigned, f)) >> 16);
}
// Pack two f32 -> packed bf16x2 dword with one v_perm_b32.
__device__ inline unsigned packbf(float a, float b) {
  unsigned ua = bfround(__builtin_bit_cast(unsigned, a));
  unsigned ub = bfround(__builtin_bit_cast(unsigned, b));
  return __builtin_amdgcn_perm(ub, ua, 0x07060302u);      // [ub.hi16 : ua.hi16]
}
__device__ inline float sigf(float x) { return 1.0f / (1.0f + __expf(-x)); }

__device__ inline v8f wmma_bf16(v16bf a, v16bf b, v8f c) {
  return __builtin_amdgcn_wmma_f32_16x16x32_bf16(false, a, false, b, (short)0, c, false, false);
}

// A fragment (16x32 bf16) from row-major bf16 matrix, row stride `ld`.
// lane: m = lane&15, half = lane>>4; V0..3: K = k0+8*half+0..7, V4..7: K = k0+16+8*half+0..7.
__device__ inline v16bf load_a_bf16(const unsigned short* base, int row, int ld, int k0, int half) {
  BF16Frag f;
  const unsigned short* p = base + (size_t)row * ld + k0 + 8 * half;
  f.q[0] = *(const v4u*)p;
  f.q[1] = *(const v4u*)(p + 16);
  return f.v;
}
// A fragment from f32 source row (convert on the fly; VALU co-executes with WMMA pipe).
__device__ inline v16bf load_a_f32(const float* rowp, int k0, int half) {
  BF16Frag f;
  const float* p0 = rowp + k0 + 8 * half;
  f.u[0] = packbf(p0[0], p0[1]);  f.u[1] = packbf(p0[2], p0[3]);
  f.u[2] = packbf(p0[4], p0[5]);  f.u[3] = packbf(p0[6], p0[7]);
  const float* p1 = p0 + 16;
  f.u[4] = packbf(p1[0], p1[1]);  f.u[5] = packbf(p1[2], p1[3]);
  f.u[6] = packbf(p1[4], p1[5]);  f.u[7] = packbf(p1[6], p1[7]);
  return f.v;
}
// B fragment (32x16 bf16): lane n = lane&15, khalf = lane>>4; K = k0+8*khalf+0..7, k0+16+8*khalf+0..7.
__device__ inline v16bf load_b_lds(const unsigned short* wl, int row, int k0, int khalf) {
  BF16Frag f;
  const unsigned short* p = wl + (size_t)row * KP + k0 + 8 * khalf;
  f.q[0] = *(const v4u*)p;
  f.q[1] = *(const v4u*)(p + 16);
  return f.v;
}

// ---------------- Device-wide barrier (persistent kernel) ----------------
__device__ inline void grid_sync(unsigned* cnt, unsigned* gen, unsigned nblk) {
  __syncthreads();
  if (threadIdx.x == 0) {
    __threadfence();
    unsigned g = __hip_atomic_load(gen, __ATOMIC_RELAXED, __HIP_MEMORY_SCOPE_AGENT);
    unsigned a = __hip_atomic_fetch_add(cnt, 1u, __ATOMIC_ACQ_REL, __HIP_MEMORY_SCOPE_AGENT);
    if (a == nblk - 1u) {
      __hip_atomic_store(cnt, 0u, __ATOMIC_RELAXED, __HIP_MEMORY_SCOPE_AGENT);
      __hip_atomic_store(gen, g + 1u, __ATOMIC_RELEASE, __HIP_MEMORY_SCOPE_AGENT);
    } else {
      while (__hip_atomic_load(gen, __ATOMIC_ACQUIRE, __HIP_MEMORY_SCOPE_AGENT) == g)
        __builtin_amdgcn_s_sleep(2);
    }
  }
  __syncthreads();
}

// ---------------- Prep kernels ----------------
__global__ void init_state(unsigned char* ws) {
  unsigned* p = (unsigned*)ws;
  int n = OFF_WP / 4;            // barrier + h_f32 + both h_bf16 buffers
  for (int i = blockIdx.x * blockDim.x + threadIdx.x; i < n; i += gridDim.x * blockDim.x)
    p[i] = 0u;
}

__global__ void pack_w(const float* __restrict__ w_ih, const float* __restrict__ w_hh,
                       unsigned short* __restrict__ wp) {
  const int total = NBLK * WSLAB;
  for (int i = blockIdx.x * blockDim.x + threadIdx.x; i < total; i += gridDim.x * blockDim.x) {
    int c   = i % KP;
    int rr  = i / KP;
    int row = rr % ROWS;
    int blk = rr / ROWS;
    int g = row >> 4, j = row & 15;
    int grow = g * H_ + blk * 16 + j;           // row in [3H, K] weight space
    float v = 0.0f;
    if (c < H_)       v = w_hh[(size_t)grow * H_ + c];
    else if (c < KC)  v = w_ih[(size_t)grow * I_ + (c - H_)];
    wp[i] = f2bf(v);
  }
}

__global__ void pack_wout(const float* __restrict__ w, unsigned short* __restrict__ o) {
  const int total = O_ * H_;
  for (int i = blockIdx.x * blockDim.x + threadIdx.x; i < total; i += gridDim.x * blockDim.x)
    o[i] = f2bf(w[i]);
}

// ---------------- Persistent GRU scan ----------------
// Grid: 64 blocks x 256 threads (8 waves). Block k owns h columns [16k, 16k+16).
// Waves 0..3 (ksel=0): batch tiles m=0..3, K in [0,768)   -> h-region only.
// Waves 4..7 (ksel=1): same tiles,      K in [768,1536)   -> h-tail + x-region.
__global__ void __launch_bounds__(256)
gru_scan(const float* __restrict__ x, const float* __restrict__ bias,
         const float* __restrict__ bn, unsigned char* __restrict__ ws) {
  extern __shared__ __align__(32) unsigned char smem[];
  unsigned short* Wlds = (unsigned short*)smem;
  v8f* red = (v8f*)(smem + WSLAB * 2);

  unsigned* bar = (unsigned*)(ws + OFF_BAR);
  float* hf = (float*)(ws + OFF_HF32);
  unsigned short* hb = (unsigned short*)(ws + OFF_HBF);
  const unsigned short* slab = (const unsigned short*)(ws + OFF_WP) + (size_t)blockIdx.x * WSLAB;

  // ---- Load this block's weight slab (48 x 1544 bf16 = 144.75 KB) into LDS via TDM ----
#if __has_builtin(__builtin_amdgcn_tensor_load_to_lds)
  if (threadIdx.x < 32) {
    unsigned lds_base = (unsigned)(size_t)Wlds;              // flat addr low 32 = LDS offset
    unsigned long long ga = (unsigned long long)(size_t)slab;
    v4u g0 = { 1u,                                           // count=1
               lds_base,                                     // bits 63:32  lds_addr
               (unsigned)(ga & 0xFFFFFFFFu),                 // bits 95:64  global_addr lo
               (unsigned)((ga >> 32) & 0x1FFFFFFu) | (2u << 30) }; // addr[56:32] | type=2
    v8i g1 = { (int)(1u << 16),                              // data_size = 1 (2 bytes)
               (int)((KP & 0xFFFF) << 16),                   // tensor_dim0 lo16
               (int)((ROWS & 0xFFFF) << 16),                 // dim0 hi16=0 | tensor_dim1 lo16
               (int)((KP & 0xFFFF) << 16),                   // dim1 hi16=0 | tile_dim0
               (int)(ROWS & 0xFFFF),                         // tile_dim1 | tile_dim2=0
               (int)KP,                                      // tensor_dim0_stride lo32
               0, 0 };
    v4i gz = { 0, 0, 0, 0 };
#if defined(__clang_major__) && (__clang_major__ >= 23)
    v8i gz8 = { 0, 0, 0, 0, 0, 0, 0, 0 };
    __builtin_amdgcn_tensor_load_to_lds(g0, g1, gz, gz, gz8, 0);
#else
    __builtin_amdgcn_tensor_load_to_lds(g0, g1, gz, gz, 0);
#endif
    __builtin_amdgcn_s_wait_tensorcnt(0);
  }
#else
  for (int i = threadIdx.x; i < WSLAB / 8; i += blockDim.x)
    ((v4u*)Wlds)[i] = ((const v4u*)slab)[i];
#endif
  __syncthreads();

  const int wave = threadIdx.x >> 5;
  const int lane = threadIdx.x & 31;
  const int mgrp = wave & 3;                                  // batch tile
  const int ksel_u = __builtin_amdgcn_readfirstlane(wave >> 2); // SCALAR K-half select
  const int half = lane >> 4;
  const int nl   = lane & 15;
  const int colg = blockIdx.x * 16 + nl;                      // owned h-column

  const float b_r  = bias[colg];
  const float b_z  = bias[H_ + colg];
  const float b_n  = bias[2 * H_ + colg];
  const float bn_c = bn[colg];

  const int brow = 16 * mgrp + nl;
  const float* xbase = x + (size_t)brow * T_ * I_;

  for (int t = 0; t < T_; ++t) {
    const unsigned short* hsrc = hb + (size_t)(t & 1) * (B_ * H_);
    unsigned short*       hdst = hb + (size_t)((t + 1) & 1) * (B_ * H_);

    v8f accR  = {0,0,0,0,0,0,0,0};
    v8f accZ  = {0,0,0,0,0,0,0,0};
    v8f accNh = {0,0,0,0,0,0,0,0};
    v8f accNx = {0,0,0,0,0,0,0,0};

    if (ksel_u == 0) {
      // K in [0, 768): all h-region. Branch-free, compile-time trips, EXEC stays full.
#pragma unroll 8
      for (int i = 0; i < 24; ++i) {
        const int kk = i * 32;
        v16bf a = load_a_bf16(hsrc, brow, H_, kk, half);
        accR  = wmma_bf16(a, load_b_lds(Wlds, 0  + nl, kk, half), accR);
        accZ  = wmma_bf16(a, load_b_lds(Wlds, 16 + nl, kk, half), accZ);
        accNh = wmma_bf16(a, load_b_lds(Wlds, 32 + nl, kk, half), accNh);
      }
    } else {
      const float* xrow = xbase + (size_t)t * I_;
      // K in [768, 1024): h-tail.
#pragma unroll 8
      for (int i = 0; i < 8; ++i) {
        const int kk = 768 + i * 32;
        v16bf a = load_a_bf16(hsrc, brow, H_, kk, half);
        accR  = wmma_bf16(a, load_b_lds(Wlds, 0  + nl, kk, half), accR);
        accZ  = wmma_bf16(a, load_b_lds(Wlds, 16 + nl, kk, half), accZ);
        accNh = wmma_bf16(a, load_b_lds(Wlds, 32 + nl, kk, half), accNh);
      }
      // K in [1024, 1536): x-region, f32 -> bf16 via v_perm (co-executes with WMMA).
#pragma unroll 8
      for (int i = 0; i < 16; ++i) {
        const int kk = 1024 + i * 32;
        v16bf a = load_a_f32(xrow, kk - 1024, half);
        accR  = wmma_bf16(a, load_b_lds(Wlds, 0  + nl, kk, half), accR);
        accZ  = wmma_bf16(a, load_b_lds(Wlds, 16 + nl, kk, half), accZ);
        accNx = wmma_bf16(a, load_b_lds(Wlds, 32 + nl, kk, half), accNx);
      }
    }

    // ---- reduce the two K halves through LDS (scalar branch on ksel) ----
    if (ksel_u == 1) {
      red[((mgrp * 4 + 0) * 32) + lane] = accR;
      red[((mgrp * 4 + 1) * 32) + lane] = accZ;
      red[((mgrp * 4 + 2) * 32) + lane] = accNh;
      red[((mgrp * 4 + 3) * 32) + lane] = accNx;
    }
    __syncthreads();

    if (ksel_u == 0) {
      accR  += red[((mgrp * 4 + 0) * 32) + lane];
      accZ  += red[((mgrp * 4 + 1) * 32) + lane];
      accNh += red[((mgrp * 4 + 2) * 32) + lane];
      accNx  = red[((mgrp * 4 + 3) * 32) + lane];   // ksel0 had no x-part

      // ---- gate math + state update for owned columns ----
#pragma unroll
      for (int v = 0; v < 8; ++v) {
        const int m = 16 * mgrp + v + 8 * half;     // batch row (C layout: VGPR v, lane half)
        const size_t idx = (size_t)m * H_ + colg;
        float r = sigf(accR[v] + b_r);
        float z = sigf(accZ[v] + b_z);
        float narg = accNx[v] + b_n + r * (accNh[v] + bn_c);
        float n = 2.0f * sigf(2.0f * narg) - 1.0f;  // tanh via fast exp
        float hold = hf[idx];
        float hnew = (1.0f - z) * n + z * hold;
        hf[idx]   = hnew;
        hdst[idx] = f2bf(hnew);
      }
    }
    grid_sync(&bar[0], &bar[1], gridDim.x);
  }
}

// ---------------- Output projection: sigmoid(h_T @ w_out^T + b_out) ----------------
__global__ void out_proj(const unsigned short* __restrict__ hbf,   // final h, bf16 [64][1024]
                         const unsigned short* __restrict__ wo,    // w_out bf16 [512][1024]
                         const float* __restrict__ b_out, float* __restrict__ out) {
  const int mt = blockIdx.x;        // 0..3  (batch tiles)
  const int nt = blockIdx.y;        // 0..31 (output tiles)
  const int lane = threadIdx.x;
  const int nl = lane & 15, half = lane >> 4;

  v8f acc = {0,0,0,0,0,0,0,0};
#pragma unroll 8
  for (int i = 0; i < 32; ++i) {
    const int kk = i * 32;
    v16bf a = load_a_bf16(hbf, 16 * mt + nl, H_, kk, half);
    BF16Frag f;
    const unsigned short* p = wo + (size_t)(nt * 16 + nl) * H_ + kk + 8 * half;
    f.q[0] = *(const v4u*)p;
    f.q[1] = *(const v4u*)(p + 16);
    acc = wmma_bf16(a, f.v, acc);
  }
  const float bo = b_out[nt * 16 + nl];
#pragma unroll
  for (int v = 0; v < 8; ++v) {
    const int m = 16 * mt + v + 8 * half;
    out[(size_t)m * O_ + nt * 16 + nl] = sigf(acc[v] + bo);
  }
}

// ---------------- Host entry ----------------
extern "C" void kernel_launch(void* const* d_in, const int* in_sizes, int n_in,
                              void* d_out, int out_size, void* d_ws, size_t ws_size,
                              hipStream_t stream) {
  const float* x     = (const float*)d_in[0];
  const float* w_ih  = (const float*)d_in[1];
  const float* w_hh  = (const float*)d_in[2];
  const float* b     = (const float*)d_in[3];
  const float* bn    = (const float*)d_in[4];
  const float* w_out = (const float*)d_in[5];
  const float* b_out = (const float*)d_in[6];
  unsigned char* ws  = (unsigned char*)d_ws;
  float* out = (float*)d_out;

  init_state<<<128, 256, 0, stream>>>(ws);
  pack_w<<<512, 256, 0, stream>>>(w_ih, w_hh, (unsigned short*)(ws + OFF_WP));
  pack_wout<<<256, 256, 0, stream>>>(w_out, (unsigned short*)(ws + OFF_WOUT));

  gru_scan<<<NBLK, 256, SMEM_BYTES, stream>>>(x, b, bn, ws);

  dim3 g(4, 32);
  out_proj<<<g, 32, 0, stream>>>((const unsigned short*)(ws + OFF_HBF),
                                 (const unsigned short*)(ws + OFF_WOUT), b_out, out);
}
